// DQRN_21431886807594
// MI455X (gfx1250) — matrix operand order repaired and checked by reading, not compiled, not run on hardware
//
#include <hip/hip_runtime.h>
#include <math.h>

// ---------------- problem constants ----------------
#define T_SEQ   256
#define BATCH   1024
#define DIN     256
#define HLOW    256
#define HHIGH   256
#define P_PAIRS 523776              // 1024*1023/2
#define PBLK    2046                // P_PAIRS / 256 exactly

typedef __attribute__((ext_vector_type(16))) __bf16 v16bf;
typedef __attribute__((ext_vector_type(8)))  __bf16 v8bf;
typedef __attribute__((ext_vector_type(8)))  float  v8f;

// Opaque zero: defeats LICM so loop-invariant global loads are re-issued
// (hitting L2) instead of being hoisted and spilled to scratch.
__device__ __forceinline__ int opaque_zero() {
    int x = 0;
    asm volatile("" : "+s"(x));
    return x;
}

// ---------------- WMMA helpers ----------------
__device__ __forceinline__ v8f wmma_bf16(v16bf a, v16bf b, v8f c) {
    // v_wmma_f32_16x16x32_bf16: D = A(16x32) * B(32x16) + C(16x16 f32)
    return __builtin_amdgcn_wmma_f32_16x16x32_bf16(
        false, a, false, b, (short)0, c, false, false);
}

// A fragment (16x32 bf16), row-major source, leading dim `ld` (elements).
// Lane l: row = l&15; elems 0..7 = K[koff..koff+7], 8..15 = K[koff+16..+23],
// koff = 8*(l>=16).  (matches ISA 16-bit A 16x32 layout)
__device__ __forceinline__ v16bf load_a_frag(const __bf16* p, int ld, int lane) {
    int row  = lane & 15;
    int koff = (lane >> 4) << 3;
    const __bf16* b0 = p + row * ld + koff;
    v8bf lo = *(const v8bf*)(b0);
    v8bf hi = *(const v8bf*)(b0 + 16);
    v16bf a;
#pragma unroll
    for (int j = 0; j < 8; ++j) { a[j] = lo[j]; a[j + 8] = hi[j]; }
    return a;
}

// B fragment (32x16 bf16) where B[k,n] = W[n, k], W row-major, leading dim ld.
// Lane l: col n = l&15; elems 0..7 = W[n, k0..k0+7], 8..15 = W[n, k0+16..+23], k0 = 8*(l>=16).
__device__ __forceinline__ v16bf load_b_frag(const __bf16* W, int ld, int lane) {
    int n  = lane & 15;
    int k0 = (lane >> 4) << 3;
    const __bf16* b0 = W + n * ld + k0;
    v8bf lo = *(const v8bf*)(b0);
    v8bf hi = *(const v8bf*)(b0 + 16);
    v16bf b;
#pragma unroll
    for (int j = 0; j < 8; ++j) { b[j] = lo[j]; b[j + 8] = hi[j]; }
    return b;
}

__device__ __forceinline__ float sigmoidf_(float x) {
    return 1.0f / (1.0f + __expf(-x));
}

// ---------------- kernel 0: fp32 -> bf16 weight staging ----------------
__global__ __launch_bounds__(256) void cvt_bf16_kernel(const float* __restrict__ a,
                                                       __bf16* __restrict__ b, int n) {
    int i = blockIdx.x * 256 + threadIdx.x;
    if (i < n) b[i] = (__bf16)a[i];
}

// ---------------- kernel 1: low GRU, persistent per batch-tile ----------------
// Grid: 64 WGs x 256 threads (8 waves). WG owns 16 batch rows for all 256 steps.
// Wave w owns hidden col-tiles {2w, 2w+1}. Per step per wave: 96 WMMAs.
// Per k-step: batch-issue 12 B-fragment loads, then 12 WMMAs -> WMMAs drain
// against in-flight loads (s_wait_loadcnt N>0) instead of load->wait->wmma.
__global__ __launch_bounds__(256) void gru_low_kernel(
    const float*  __restrict__ x,          // [T,B,DIN] fp32
    const __bf16* __restrict__ WihB,       // [3H, DIN] bf16
    const __bf16* __restrict__ WhhB,       // [3H, H] bf16
    const float*  __restrict__ bih,        // [3H]
    const float*  __restrict__ bhh,        // [3H]
    const float*  __restrict__ Wcl,        // [4, H]
    const float*  __restrict__ bcl,        // [4]
    __bf16*       __restrict__ cluster_bf, // [B, H] bf16 out
    float*        __restrict__ cluster_proj) // [B, 4] out
{
    __shared__ __bf16 sx[16 * DIN];
    __shared__ __bf16 sh_bf[2][16 * HLOW];
    __shared__ float  sh_f [2][16 * HLOW];

    const int tid     = threadIdx.x;
    const int lane    = tid & 31;
    const int wave    = tid >> 5;
    const int rowbase = blockIdx.x * 16;

    // h0 = 0 in buffer 0
    for (int i = tid; i < 16 * HLOW; i += 256) {
        sh_bf[0][i] = (__bf16)0.0f;
        sh_f[0][i]  = 0.0f;
    }
    __syncthreads();

    const int ct0   = wave * 2;           // first hidden col-tile owned by wave
    const int coll  = lane & 15;          // local output column
    const int rbase = (lane >> 4) << 3;   // 0 or 8 (C/D row offset)

    for (int t = 0; t < T_SEQ; ++t) {
        const int rd = t & 1, wr = rd ^ 1;

        // stage x_t tile into LDS as bf16 (one HBM read of x total)
        {
            const float* xp = x + ((size_t)t * BATCH + rowbase) * DIN;
            int r = tid >> 4, c0 = (tid & 15) * 16;
            const float4* src = (const float4*)(xp + r * DIN + c0);
#pragma unroll
            for (int q = 0; q < 4; ++q) {
                float4 v = src[q];
                __bf16* d = &sx[r * DIN + c0 + q * 4];
                d[0] = (__bf16)v.x; d[1] = (__bf16)v.y;
                d[2] = (__bf16)v.z; d[3] = (__bf16)v.w;
            }
        }
        __syncthreads();

        // per-iteration opaque offset: keeps weight loads inside the loop
        const int oz = opaque_zero();
        const __bf16* WihT = WihB + oz;
        const __bf16* WhhT = WhhB + oz;

        v8f aR[2]  = {v8f{}, v8f{}};
        v8f aZ[2]  = {v8f{}, v8f{}};
        v8f aXN[2] = {v8f{}, v8f{}};
        v8f aHN[2] = {v8f{}, v8f{}};

#pragma unroll 1
        for (int kt = 0; kt < 8; ++kt) {
            const int ko = kt * 32;
            const v16bf ax = load_a_frag(&sx[ko],        DIN,  lane);
            const v16bf ah = load_a_frag(&sh_bf[rd][ko], HLOW, lane);

            // batch-issue all 12 weight fragments (independent streams)
            v16bf bw[12];
#pragma unroll
            for (int ci = 0; ci < 2; ++ci) {
                const int ct = ct0 + ci;
                bw[ci * 6 + 0] = load_b_frag(WihT + (size_t)( ct       * 16) * DIN  + ko, DIN,  lane);
                bw[ci * 6 + 1] = load_b_frag(WhhT + (size_t)( ct       * 16) * HLOW + ko, HLOW, lane);
                bw[ci * 6 + 2] = load_b_frag(WihT + (size_t)((16 + ct) * 16) * DIN  + ko, DIN,  lane);
                bw[ci * 6 + 3] = load_b_frag(WhhT + (size_t)((16 + ct) * 16) * HLOW + ko, HLOW, lane);
                bw[ci * 6 + 4] = load_b_frag(WihT + (size_t)((32 + ct) * 16) * DIN  + ko, DIN,  lane);
                bw[ci * 6 + 5] = load_b_frag(WhhT + (size_t)((32 + ct) * 16) * HLOW + ko, HLOW, lane);
            }
            // then drain with 12 WMMAs
#pragma unroll
            for (int ci = 0; ci < 2; ++ci) {
                aR[ci]  = wmma_bf16(ax, bw[ci * 6 + 0], aR[ci]);
                aR[ci]  = wmma_bf16(ah, bw[ci * 6 + 1], aR[ci]);
                aZ[ci]  = wmma_bf16(ax, bw[ci * 6 + 2], aZ[ci]);
                aZ[ci]  = wmma_bf16(ah, bw[ci * 6 + 3], aZ[ci]);
                aXN[ci] = wmma_bf16(ax, bw[ci * 6 + 4], aXN[ci]);
                aHN[ci] = wmma_bf16(ah, bw[ci * 6 + 5], aHN[ci]);
            }
        }

#pragma unroll
        for (int ci = 0; ci < 2; ++ci) {
            const int ct   = ct0 + ci;
            const int gcol = ct * 16 + coll;         // hidden column 0..255
            const float br_ = bih[gcol]       + bhh[gcol];
            const float bz_ = bih[256 + gcol] + bhh[256 + gcol];
            const float bxn = bih[512 + gcol];
            const float bhn = bhh[512 + gcol];
#pragma unroll
            for (int v = 0; v < 8; ++v) {
                const int row = rbase + v;
                const float r  = sigmoidf_(aR[ci][v] + br_);
                const float z  = sigmoidf_(aZ[ci][v] + bz_);
                const float n  = tanhf(aXN[ci][v] + bxn + r * (aHN[ci][v] + bhn));
                const float ho = sh_f[rd][row * HLOW + gcol];
                const float hn = (1.0f - z) * n + z * ho;
                sh_f [wr][row * HLOW + gcol] = hn;
                sh_bf[wr][row * HLOW + gcol] = (__bf16)hn;
            }
        }
        __syncthreads();
    }

    const int fb = ((T_SEQ - 1) & 1) ^ 1;   // final h buffer (== 0 for even T)

    // export cluster_rep in bf16 for the high GRU GEMM
    for (int i = tid; i < 16 * HLOW; i += 256)
        cluster_bf[(size_t)rowbase * HLOW + i] = sh_bf[fb][i];

    // cluster projection: tanh(h @ Wcl^T + bcl) -> [16,4]
    if (tid < 64) {
        const int r = tid >> 2, g = tid & 3;
        float acc = bcl[g];
        for (int k = 0; k < HLOW; ++k)
            acc += sh_f[fb][r * HLOW + k] * Wcl[g * HLOW + k];
        cluster_proj[(size_t)(rowbase + r) * 4 + g] = tanhf(acc);
    }
}

// ---------------- kernel 2a: xp_high = cluster_bf @ W_ih_high^T + b ----------------
__global__ __launch_bounds__(256) void xp_high_kernel(
    const __bf16* __restrict__ A,    // [1024, HLOW] bf16
    const __bf16* __restrict__ W,    // [3H, HLOW] bf16
    const float*  __restrict__ bih,  // [3H]
    float*        __restrict__ xp)   // [1024, 3H]
{
    const int lane    = threadIdx.x & 31;
    const int wave    = threadIdx.x >> 5;
    const int rowbase = blockIdx.x * 16;

    const int coll  = lane & 15;
    const int rbase = (lane >> 4) << 3;

    // A fragments are reused across all 6 n-tiles: cache them (64 VGPRs)
    v16bf af[8];
#pragma unroll
    for (int kt = 0; kt < 8; ++kt)
        af[kt] = load_a_frag(A + (size_t)rowbase * HLOW + kt * 32, HLOW, lane);

#pragma unroll 1
    for (int j = 0; j < 6; ++j) {
        const int nt = wave * 6 + j;             // 0..47
        // batch-issue the 8 weight fragments, then drain 8 WMMAs
        v16bf bw[8];
#pragma unroll
        for (int kt = 0; kt < 8; ++kt)
            bw[kt] = load_b_frag(W + (size_t)(nt * 16) * HLOW + kt * 32, HLOW, lane);
        v8f acc = {};
#pragma unroll
        for (int kt = 0; kt < 8; ++kt)
            acc = wmma_bf16(af[kt], bw[kt], acc);

        const int gcol = nt * 16 + coll;
        const float b  = bih[gcol];
#pragma unroll
        for (int v = 0; v < 8; ++v)
            xp[(size_t)(rowbase + rbase + v) * (3 * HHIGH) + gcol] = acc[v] + b;
    }
}

// ---------------- kernel 2b: high GRU recurrence (batch 1, serial tail) ----------
__global__ __launch_bounds__(256) void gru_high_kernel(
    const float* __restrict__ xp,    // [1024, 3H] (includes b_ih)
    const float* __restrict__ Whh,   // [3H, H] fp32
    const float* __restrict__ bhh,   // [3H]
    const float* __restrict__ Wst,   // [4, H]
    const float* __restrict__ bst,   // [4]
    const float* __restrict__ Wa1,   // [4, 8]
    const float* __restrict__ ba1,   // [4]
    float*       __restrict__ s4out) // [4] : b_a1 + W_a1[:,0:4] @ state_rep
{
    __shared__ float h[HHIGH];
    __shared__ float sp[4];
    const int tid = threadIdx.x;
    h[tid] = 0.0f;
    __syncthreads();

    const float br_ = bhh[tid], bz_ = bhh[HHIGH + tid], bn_ = bhh[2 * HHIGH + tid];

    for (int s = 0; s < BATCH; ++s) {
        const int oz = opaque_zero();   // keep Whh loads in-loop (L2 hits)
        const float* wr_ = Whh + (size_t)tid * HHIGH + oz;
        const float* wz_ = Whh + (size_t)(HHIGH + tid) * HHIGH + oz;
        const float* wn_ = Whh + (size_t)(2 * HHIGH + tid) * HHIGH + oz;

        float gr = 0.0f, gz = 0.0f, gn = 0.0f;
#pragma unroll 4
        for (int k = 0; k < HHIGH; ++k) {
            const float hv = h[k];
            gr = fmaf(hv, wr_[k], gr);
            gz = fmaf(hv, wz_[k], gz);
            gn = fmaf(hv, wn_[k], gn);
        }
        const float* xs = xp + (size_t)s * (3 * HHIGH);
        const float r = sigmoidf_(xs[tid] + gr + br_);
        const float z = sigmoidf_(xs[HHIGH + tid] + gz + bz_);
        const float n = tanhf(xs[2 * HHIGH + tid] + r * (gn + bn_));
        const float hn = (1.0f - z) * n + z * h[tid];
        __syncthreads();
        h[tid] = hn;
        __syncthreads();
    }

    if (tid < 4) {
        float acc = bst[tid];
        for (int k = 0; k < HHIGH; ++k) acc += h[k] * Wst[tid * HHIGH + k];
        sp[tid] = tanhf(acc);
    }
    __syncthreads();
    if (tid < 4) {
        float a = ba1[tid];
#pragma unroll
        for (int k = 0; k < 4; ++k) a += Wa1[tid * 8 + k] * sp[k];
        s4out[tid] = a;
    }
}

// ---------------- kernel 3: pairwise q + block max ----------------
__global__ __launch_bounds__(256) void pair_q_kernel(
    const float* __restrict__ cp,   // [1024,4] cluster_proj
    const float* __restrict__ s4,   // [4]
    const float* __restrict__ Wa1,  // [4,8]
    const float* __restrict__ Wa2,  // [1,4]
    const float* __restrict__ ba2,  // [1]
    float*       __restrict__ q,    // [P]
    float*       __restrict__ pmax) // [PBLK]
{
    const int p = blockIdx.x * 256 + threadIdx.x;
    float qv = -3.402823466e38f;
    if (p < P_PAIRS) {
        // invert row-major tril(k=-1) index: pairs (i, j<i)
        int i = (int)((1.0f + sqrtf(1.0f + 8.0f * (float)p)) * 0.5f);
        while (i * (i - 1) / 2 > p) --i;
        while ((i + 1) * i / 2 <= p) ++i;
        const int j = p - i * (i - 1) / 2;

        const float m0 = cp[i * 4 + 0] + cp[j * 4 + 0];
        const float m1 = cp[i * 4 + 1] + cp[j * 4 + 1];
        const float m2 = cp[i * 4 + 2] + cp[j * 4 + 2];
        const float m3 = cp[i * 4 + 3] + cp[j * 4 + 3];

        float acc = ba2[0];
#pragma unroll
        for (int g = 0; g < 4; ++g) {
            float hg = s4[g] + Wa1[g * 8 + 4] * m0 + Wa1[g * 8 + 5] * m1
                             + Wa1[g * 8 + 6] * m2 + Wa1[g * 8 + 7] * m3;
            hg = fmaxf(hg, 0.0f);
            acc = fmaf(Wa2[g], hg, acc);
        }
        qv = acc;
        q[p] = qv;
    }
    __shared__ float red[256];
    red[threadIdx.x] = qv;
    __syncthreads();
#pragma unroll
    for (int s = 128; s > 0; s >>= 1) {
        if (threadIdx.x < s) red[threadIdx.x] = fmaxf(red[threadIdx.x], red[threadIdx.x + s]);
        __syncthreads();
    }
    if (threadIdx.x == 0) pmax[blockIdx.x] = red[0];
}

// ---------------- kernels 4-7: softmax reductions ----------------
__global__ __launch_bounds__(256) void reduce_max_kernel(const float* __restrict__ pmax,
                                                         int n, float* __restrict__ out) {
    __shared__ float red[256];
    float m = -3.402823466e38f;
    for (int i = threadIdx.x; i < n; i += 256) m = fmaxf(m, pmax[i]);
    red[threadIdx.x] = m;
    __syncthreads();
#pragma unroll
    for (int s = 128; s > 0; s >>= 1) {
        if (threadIdx.x < s) red[threadIdx.x] = fmaxf(red[threadIdx.x], red[threadIdx.x + s]);
        __syncthreads();
    }
    if (threadIdx.x == 0) out[0] = red[0];
}

__global__ __launch_bounds__(256) void exp_psum_kernel(float* __restrict__ q,
                                                       const float* __restrict__ gmax,
                                                       float* __restrict__ psum) {
    const int p = blockIdx.x * 256 + threadIdx.x;
    const float M = gmax[0];
    float e = 0.0f;
    if (p < P_PAIRS) { e = __expf(q[p] - M); q[p] = e; }
    __shared__ float red[256];
    red[threadIdx.x] = e;
    __syncthreads();
#pragma unroll
    for (int s = 128; s > 0; s >>= 1) {
        if (threadIdx.x < s) red[threadIdx.x] += red[threadIdx.x + s];
        __syncthreads();
    }
    if (threadIdx.x == 0) psum[blockIdx.x] = red[0];
}

__global__ __launch_bounds__(256) void reduce_sum_inv_kernel(const float* __restrict__ psum,
                                                             int n, float* __restrict__ out) {
    __shared__ float red[256];
    float s = 0.0f;
    for (int i = threadIdx.x; i < n; i += 256) s += psum[i];
    red[threadIdx.x] = s;
    __syncthreads();
#pragma unroll
    for (int t = 128; t > 0; t >>= 1) {
        if (threadIdx.x < t) red[threadIdx.x] += red[threadIdx.x + t];
        __syncthreads();
    }
    if (threadIdx.x == 0) out[0] = 1.0f / red[0];
}

__global__ __launch_bounds__(256) void scale_kernel(float* __restrict__ q,
                                                    const float* __restrict__ inv) {
    const int p = blockIdx.x * 256 + threadIdx.x;
    if (p < P_PAIRS) q[p] *= inv[0];
}

// ---------------- host launch ----------------
extern "C" void kernel_launch(void* const* d_in, const int* in_sizes, int n_in,
                              void* d_out, int out_size, void* d_ws, size_t ws_size,
                              hipStream_t stream) {
    const float* x      = (const float*)d_in[0];
    const float* Wih_l  = (const float*)d_in[1];
    const float* Whh_l  = (const float*)d_in[2];
    const float* bih_l  = (const float*)d_in[3];
    const float* bhh_l  = (const float*)d_in[4];
    const float* Wih_h  = (const float*)d_in[5];
    const float* Whh_h  = (const float*)d_in[6];
    const float* bih_h  = (const float*)d_in[7];
    const float* bhh_h  = (const float*)d_in[8];
    const float* Wst    = (const float*)d_in[9];
    const float* bst    = (const float*)d_in[10];
    const float* Wcl    = (const float*)d_in[11];
    const float* bcl    = (const float*)d_in[12];
    const float* Wa1    = (const float*)d_in[13];
    const float* ba1    = (const float*)d_in[14];
    const float* Wa2    = (const float*)d_in[15];
    const float* ba2    = (const float*)d_in[16];

    char* ws = (char*)d_ws;
    size_t off = 0;
    auto take = [&](size_t bytes) -> char* {
        char* p = ws + off;
        off = (off + bytes + 255) & ~(size_t)255;
        return p;
    };
    const int WN = 3 * HLOW * DIN; // 196608 elements per weight matrix

    __bf16* wihl_bf    = (__bf16*)take((size_t)WN * 2);
    __bf16* whhl_bf    = (__bf16*)take((size_t)WN * 2);
    __bf16* wihh_bf    = (__bf16*)take((size_t)WN * 2);
    __bf16* cluster_bf = (__bf16*)take((size_t)BATCH * HLOW * 2);
    float*  cl_proj    = (float*) take((size_t)BATCH * 4 * 4);
    float*  xp_high    = (float*) take((size_t)BATCH * 3 * HHIGH * 4);
    float*  s4         = (float*) take(4 * 4);
    float*  red        = (float*) take(2 * 4);
    float*  pmax       = (float*) take((size_t)PBLK * 4);
    float*  psum       = (float*) take((size_t)PBLK * 4);
    float*  q          = (float*)d_out;

    // stage weights to bf16
    cvt_bf16_kernel<<<(WN + 255) / 256, 256, 0, stream>>>(Wih_l, wihl_bf, WN);
    cvt_bf16_kernel<<<(WN + 255) / 256, 256, 0, stream>>>(Whh_l, whhl_bf, WN);
    cvt_bf16_kernel<<<(WN + 255) / 256, 256, 0, stream>>>(Wih_h, wihh_bf, WN);

    // low GRU (dominant WMMA kernel): 64 WGs x 16 batch rows, persistent over T
    gru_low_kernel<<<BATCH / 16, 256, 0, stream>>>(
        x, wihl_bf, whhl_bf, bih_l, bhh_l, Wcl, bcl, cluster_bf, cl_proj);

    // high GRU input GEMM (WMMA) + serial recurrence
    xp_high_kernel<<<BATCH / 16, 256, 0, stream>>>(cluster_bf, wihh_bf, bih_h, xp_high);
    gru_high_kernel<<<1, 256, 0, stream>>>(xp_high, Whh_h, bhh_h, Wst, bst, Wa1, ba1, s4);

    // pairwise scores + softmax
    pair_q_kernel<<<PBLK, 256, 0, stream>>>(cl_proj, s4, Wa1, Wa2, ba2, q, pmax);
    reduce_max_kernel<<<1, 256, 0, stream>>>(pmax, PBLK, red);
    exp_psum_kernel<<<PBLK, 256, 0, stream>>>(q, red, psum);
    reduce_sum_inv_kernel<<<1, 256, 0, stream>>>(psum, PBLK, red + 1);
    scale_kernel<<<PBLK, 256, 0, stream>>>(q, red + 1);

    (void)in_sizes; (void)n_in; (void)out_size; (void)ws_size;
}
